// PlasticMultiHeadAttention_90366111908076
// MI455X (gfx1250) — matrix operand order repaired and checked by reading, compiled, not run on hardware
//
#include <hip/hip_runtime.h>

// ---------------------------------------------------------------------------
// PlasticMultiHeadAttention for gfx1250 (MI455X), bf16 WMMA pipeline.
//   B=1, S=4096, E=1024, H=16, D=64
// ws layout (bytes):
//   [0      , 8Mi )  Xbf   [4096][1024] bf16
//   [8Mi    , 14Mi)  Wbf   [3072][1024] bf16
//   [14Mi   , 22Mi)  Qbf   [16][4096][64] bf16  (pre-scaled by 1/8)
//   [22Mi   , 30Mi)  Kbf   [16][4096][64] bf16
//   [30Mi   , 38Mi)  Vt    [16][64][4096] bf16  (transposed for P@V B-operand)
// ---------------------------------------------------------------------------

typedef __bf16 bf16_t;
typedef __attribute__((ext_vector_type(16))) __bf16 v16bf;
typedef __attribute__((ext_vector_type(8)))  __bf16 v8bf;
typedef __attribute__((ext_vector_type(8)))  float  v8f;

#define SEQ   4096
#define EMB   1024
#define NH    16
#define HD    64
#define E3    3072

__device__ __forceinline__ v8f wmma_bf16(v16bf a, v16bf b, v8f c) {
  return __builtin_amdgcn_wmma_f32_16x16x32_bf16(
      /*neg_a=*/false, a, /*neg_b=*/false, b,
      /*c_mod=*/(short)0, c, /*reuse_a=*/false, /*reuse_b=*/false);
}

// A-operand fragment (16x32, row-major source). lanehi = lane>>4.
// Low-half lanes hold K {base..+7, base+16..+23}, high half {+8..+15, +24..+31}.
__device__ __forceinline__ v16bf load_a(const bf16_t* rowp, int lanehi) {
  const bf16_t* p = rowp + lanehi * 8;
  v8bf lo = *(const v8bf*)(p);
  v8bf hi = *(const v8bf*)(p + 16);
  v16bf r;
#pragma unroll
  for (int i = 0; i < 8; ++i) { r[i] = lo[i]; r[i + 8] = hi[i]; }
  return r;
}

// B-operand fragment (32x16, one "column" stored K-major).
// Low-half lanes hold K {0..7, 16..23}, high half {8..15, 24..31}.
__device__ __forceinline__ v16bf load_b(const bf16_t* colp, int lanehi) {
  const bf16_t* p = colp + lanehi * 8;
  v8bf lo = *(const v8bf*)(p);
  v8bf hi = *(const v8bf*)(p + 16);
  v16bf r;
#pragma unroll
  for (int i = 0; i < 8; ++i) { r[i] = lo[i]; r[i + 8] = hi[i]; }
  return r;
}

// ---------------------------------------------------------------------------
// fp32 -> bf16 conversion, 8 elems/thread, 16B stores.
// ---------------------------------------------------------------------------
__global__ __launch_bounds__(256) void cvt_bf16(const float* __restrict__ in,
                                                bf16_t* __restrict__ out, int n8) {
  int i = blockIdx.x * blockDim.x + threadIdx.x;
  if (i >= n8) return;
  const float4* p = (const float4*)in + (size_t)i * 2;
  float4 a = p[0];
  float4 b = p[1];
  v8bf o;
  o[0] = (bf16_t)a.x; o[1] = (bf16_t)a.y; o[2] = (bf16_t)a.z; o[3] = (bf16_t)a.w;
  o[4] = (bf16_t)b.x; o[5] = (bf16_t)b.y; o[6] = (bf16_t)b.z; o[7] = (bf16_t)b.w;
  ((v8bf*)out)[i] = o;
}

// ---------------------------------------------------------------------------
// QKV GEMM: C[m,n] = sum_k X[m,k] * W[n,k] + b[n].  One wave -> 32x64 slab
// (2 m-tiles x 4 n-tiles, 8 accumulators): each B fragment feeds 2 WMMAs,
// each A fragment 4, so loads/WMMA = 1.5.  Double-buffered k-loop.
// Tasks = 128 mtiles * 48 ngroups = 6144 waves = 768 blocks * 8 waves.
// ---------------------------------------------------------------------------
__global__ __launch_bounds__(256) void qkv_gemm(
    const bf16_t* __restrict__ X, const bf16_t* __restrict__ W,
    const float* __restrict__ bias,
    bf16_t* __restrict__ Qb, bf16_t* __restrict__ Kb, bf16_t* __restrict__ Vt) {
  const int lane   = threadIdx.x & 31;
  const int wave   = threadIdx.x >> 5;
  const int task   = blockIdx.x * 8 + wave;
  const int mt     = task / (E3 / 64);
  const int ng     = task % (E3 / 64);
  const int m0     = mt * 32;
  const int n0     = ng * 64;
  const int row    = lane & 15;
  const int lanehi = lane >> 4;

  const bf16_t* Arow[2];
#pragma unroll
  for (int i = 0; i < 2; ++i) Arow[i] = X + (size_t)(m0 + i * 16 + row) * EMB;
  const bf16_t* Brow[4];
#pragma unroll
  for (int j = 0; j < 4; ++j) Brow[j] = W + (size_t)(n0 + j * 16 + row) * EMB;

  v8f acc[8];
#pragma unroll
  for (int i = 0; i < 8; ++i) acc[i] = v8f{};

  v16bf a_cur[2], b_cur[4];
#pragma unroll
  for (int i = 0; i < 2; ++i) a_cur[i] = load_a(Arow[i], lanehi);
#pragma unroll
  for (int j = 0; j < 4; ++j) b_cur[j] = load_b(Brow[j], lanehi);

#pragma unroll 2
  for (int k0 = 0; k0 < EMB; k0 += 32) {
    const int k1 = (k0 + 32) & (EMB - 1);   // wraps to 0 on last iter (discarded)
    v16bf a_nxt[2], b_nxt[4];
#pragma unroll
    for (int i = 0; i < 2; ++i) a_nxt[i] = load_a(Arow[i] + k1, lanehi);
#pragma unroll
    for (int j = 0; j < 4; ++j) b_nxt[j] = load_b(Brow[j] + k1, lanehi);

#pragma unroll
    for (int i = 0; i < 2; ++i)
#pragma unroll
      for (int j = 0; j < 4; ++j)
        acc[i * 4 + j] = wmma_bf16(a_cur[i], b_cur[j], acc[i * 4 + j]);

#pragma unroll
    for (int i = 0; i < 2; ++i) a_cur[i] = a_nxt[i];
#pragma unroll
    for (int j = 0; j < 4; ++j) b_cur[j] = b_nxt[j];
  }

  const int mlo = (lane < 16) ? 0 : 8;
#pragma unroll
  for (int j = 0; j < 4; ++j) {
    const int n   = n0 + j * 16 + row;
    const float bn = bias[n];
    const int seg = n >> 10;       // 0=Q, 1=K, 2=V
    const int nn  = n & 1023;
    const int h   = nn >> 6;
    const int d   = nn & 63;
#pragma unroll
    for (int i = 0; i < 2; ++i) {
#pragma unroll
      for (int r = 0; r < 8; ++r) {
        const int m = m0 + i * 16 + mlo + r;
        const float v = acc[i * 4 + j][r] + bn;
        if (seg == 0)      Qb[((size_t)h * SEQ + m) * HD + d] = (bf16_t)(v * 0.125f);
        else if (seg == 1) Kb[((size_t)h * SEQ + m) * HD + d] = (bf16_t)v;
        else               Vt[((size_t)h * HD + d) * SEQ + m] = (bf16_t)v;
      }
    }
  }
}

// ---------------------------------------------------------------------------
// Flash attention: one wave = (head, 16-query tile). 64 iters of 64 keys.
// 16 WMMAs per softmax pass; V fragments loaded before the softmax VALU
// section so they are in flight during exp/shuffle work.
// Tasks = 16*256 = 4096 waves = 512 blocks * 8 waves.  LDS: 8 * 2KB.
// ---------------------------------------------------------------------------
__global__ __launch_bounds__(256) void attn(
    const bf16_t* __restrict__ Qb, const bf16_t* __restrict__ Kb,
    const bf16_t* __restrict__ Vt, const float* __restrict__ trace,
    float* __restrict__ out) {
  __shared__ __align__(16) bf16_t pS[8][16 * 64];

  const int lane   = threadIdx.x & 31;
  const int wave   = threadIdx.x >> 5;
  const int task   = blockIdx.x * 8 + wave;
  const int h      = task >> 8;
  const int qt     = task & 255;
  const int q0     = qt * 16;
  const int row    = lane & 15;
  const int lanehi = lane >> 4;
  const int mlo    = (lane < 16) ? 0 : 8;

  const bf16_t* Qh = Qb + (size_t)h * SEQ * HD;
  const bf16_t* Kh = Kb + (size_t)h * SEQ * HD;
  const bf16_t* Vh = Vt + (size_t)h * HD * SEQ;
  bf16_t* myP = pS[wave];

  // Q fragments held for the whole loop (D=64 -> two K=32 chunks).
  const v16bf qf0 = load_a(Qh + (size_t)(q0 + row) * HD, lanehi);
  const v16bf qf1 = load_a(Qh + (size_t)(q0 + row) * HD + 32, lanehi);

  v8f o[4] = {v8f{}, v8f{}, v8f{}, v8f{}};
  float mrun[8], lrun[8];
#pragma unroll
  for (int r = 0; r < 8; ++r) { mrun[r] = -1e30f; lrun[r] = 0.f; }

  for (int kb = 0; kb < SEQ; kb += 64) {
    // ---- scores: four 16-key subtiles, D split 0..31 / 32..63 (8 WMMAs) ----
    v8f s[4];
#pragma unroll
    for (int kt = 0; kt < 4; ++kt) {
      const bf16_t* krow = Kh + (size_t)(kb + kt * 16 + row) * HD;
      v16bf kf = load_b(krow, lanehi);
      v8f acc = {};
      acc = wmma_bf16(qf0, kf, acc);
      kf = load_b(krow + 32, lanehi);
      s[kt] = wmma_bf16(qf1, kf, acc);
    }

    // ---- issue V fragment loads now; consumed after softmax (overlap) ----
    v16bf vf[8];
#pragma unroll
    for (int dt = 0; dt < 4; ++dt) {
      const bf16_t* vrow = Vh + (size_t)(dt * 16 + row) * SEQ + kb;
      vf[2 * dt]     = load_b(vrow, lanehi);
      vf[2 * dt + 1] = load_b(vrow + 32, lanehi);
    }

    // ---- prefetch next key block's K and V lines (global_prefetch_b8) ----
    {
      const int kbn = (kb + 64) & (SEQ - 1);
      __builtin_prefetch(Kh + (size_t)(kbn + lane) * HD, 0, 1);
      __builtin_prefetch(Kh + (size_t)(kbn + 32 + lane) * HD, 0, 1);
      __builtin_prefetch(Vh + (size_t)lane * SEQ + kbn, 0, 1);
      __builtin_prefetch(Vh + (size_t)(32 + lane) * SEQ + kbn, 0, 1);
    }

    // ---- online softmax over 64 keys ----
    asm volatile("" ::: "memory");  // keep LDS stores after last iter's loads
#pragma unroll
    for (int r = 0; r < 8; ++r) {
      float v = fmaxf(fmaxf(s[0][r], s[1][r]), fmaxf(s[2][r], s[3][r]));
      v = fmaxf(v, __shfl_xor(v, 1, 32));
      v = fmaxf(v, __shfl_xor(v, 2, 32));
      v = fmaxf(v, __shfl_xor(v, 4, 32));
      v = fmaxf(v, __shfl_xor(v, 8, 32));
      const float mnew  = fmaxf(mrun[r], v);
      const float scale = __expf(mrun[r] - mnew);
      mrun[r] = mnew;
      float p[4], ps = 0.f;
#pragma unroll
      for (int kt = 0; kt < 4; ++kt) { p[kt] = __expf(s[kt][r] - mnew); ps += p[kt]; }
      ps += __shfl_xor(ps, 1, 32);
      ps += __shfl_xor(ps, 2, 32);
      ps += __shfl_xor(ps, 4, 32);
      ps += __shfl_xor(ps, 8, 32);
      lrun[r] = lrun[r] * scale + ps;
#pragma unroll
      for (int dt = 0; dt < 4; ++dt) o[dt][r] *= scale;
      const int m = mlo + r;
#pragma unroll
      for (int kt = 0; kt < 4; ++kt)
        myP[m * 64 + kt * 16 + row] = (bf16_t)p[kt];
    }

    // ---- LDS round-trip: reshape P (C-layout) -> two A-operand fragments ----
    asm volatile("s_wait_dscnt 0" ::: "memory");
    const v16bf pf0 = load_a(myP + row * 64, lanehi);
    const v16bf pf1 = load_a(myP + row * 64 + 32, lanehi);

    // ---- O += P @ V  (8 WMMAs) ----
#pragma unroll
    for (int dt = 0; dt < 4; ++dt) {
      o[dt] = wmma_bf16(pf0, vf[2 * dt], o[dt]);
      o[dt] = wmma_bf16(pf1, vf[2 * dt + 1], o[dt]);
    }
  }

  // epilogue: normalize, apply hebbian diagonal, write fp32 output
  float dg[4];
#pragma unroll
  for (int dt = 0; dt < 4; ++dt) {
    const int d = dt * 16 + row;
    dg[dt] = trace[(size_t)h * HD * HD + (size_t)d * (HD + 1)];
  }
#pragma unroll
  for (int r = 0; r < 8; ++r) {
    const float inv = 1.0f / lrun[r];
    const int q = q0 + mlo + r;
#pragma unroll
    for (int dt = 0; dt < 4; ++dt) {
      const int d = dt * 16 + row;
      out[(size_t)q * EMB + h * HD + d] = o[dt][r] * inv * dg[dt];
    }
  }
}

// ---------------------------------------------------------------------------
extern "C" void kernel_launch(void* const* d_in, const int* in_sizes, int n_in,
                              void* d_out, int out_size, void* d_ws, size_t ws_size,
                              hipStream_t stream) {
  const float* hs = (const float*)d_in[0];   // [1,4096,1024]
  const float* Wq = (const float*)d_in[1];   // [3072,1024]
  const float* bq = (const float*)d_in[2];   // [3072]
  const float* tr = (const float*)d_in[3];   // [16,64,64]
  float* out = (float*)d_out;

  char* ws = (char*)d_ws;
  bf16_t* Xb = (bf16_t*)(ws);
  bf16_t* Wb = (bf16_t*)(ws + (size_t)8  * 1024 * 1024);
  bf16_t* Qb = (bf16_t*)(ws + (size_t)14 * 1024 * 1024);
  bf16_t* Kb = (bf16_t*)(ws + (size_t)22 * 1024 * 1024);
  bf16_t* Vt = (bf16_t*)(ws + (size_t)30 * 1024 * 1024);

  {
    int n8 = (SEQ * EMB) / 8;                 // 524288
    cvt_bf16<<<(n8 + 255) / 256, 256, 0, stream>>>(hs, Xb, n8);
  }
  {
    int n8 = (E3 * EMB) / 8;                  // 393216
    cvt_bf16<<<(n8 + 255) / 256, 256, 0, stream>>>(Wq, Wb, n8);
  }
  // 6144 wave tasks / 8 waves per block
  qkv_gemm<<<768, 256, 0, stream>>>(Xb, Wb, bq, Qb, Kb, Vt);
  // 4096 wave tasks / 8 waves per block
  attn<<<512, 256, 0, stream>>>(Qb, Kb, Vt, tr, out);
}
